// NerfTriplane_79164837200477
// MI455X (gfx1250) — compile-verified, matrix operand order checked by reading, not run on hardware
//
#include <hip/hip_runtime.h>
#include <cstdint>

typedef __attribute__((ext_vector_type(16))) _Float16 v16h;
typedef __attribute__((ext_vector_type(8)))  _Float16 v8h;
typedef __attribute__((ext_vector_type(8)))  float    v8f;

union FragU {
    v16h h;
    unsigned int u[8];
    uint4 q[2];
};

union H8 {
    uint4 q;
    v8h v;
};

__device__ __forceinline__ int iclamp(int v, int lo, int hi) {
    return v < lo ? lo : (v > hi ? hi : v);
}

__device__ __forceinline__ v8f splat8(float v) {
    v8f r;
#pragma unroll
    for (int i = 0; i < 8; ++i) r[i] = v;
    return r;
}

__device__ __forceinline__ v8h splath(_Float16 v) {
    v8h r;
#pragma unroll
    for (int i = 0; i < 8; ++i) r[i] = v;
    return r;
}

__device__ __forceinline__ v8f lrelu8(v8f v) {
#pragma unroll
    for (int i = 0; i < 8; ++i) {
        float x = v[i];
        v[i] = x >= 0.0f ? x : 0.01f * x;
    }
    return v;
}

__device__ __forceinline__ v8f wmma_f16(const v16h a, const v16h b, v8f c) {
    return __builtin_amdgcn_wmma_f32_16x16x32_f16(false, a, false, b, (short)0, c, false, false);
}

// A fragment (16x32 f16) load from row-major [16][128] f16 staging in LDS.
// Lane L (L<16): row M=L, K=0..7 in v0-3 (pairs), K=16..23 in v4-7.
// Lane L+16:     row M=L, K=8..15 / K=24..31.  (ISA 7.12.2, 16-bit A 16x32)
__device__ __forceinline__ void loadA(FragU& a, const _Float16* tile_base, int kt, int lane) {
    const unsigned int* sp = (const unsigned int*)tile_base; // pair view, row stride 64 pairs
    int m  = lane & 15;
    int kh = (lane >> 4) * 4;
#pragma unroll
    for (int v = 0; v < 8; ++v) {
        int kp = kt * 16 + (v < 4 ? 0 : 8) + kh + (v & 3);
        a.u[v] = sp[m * 64 + kp];
    }
}

// B fragment from pre-packed fragment-major weights (512 halves / frag, 16 per lane).
__device__ __forceinline__ void loadB(FragU& b, const _Float16* wfrag, int fragIdx, int lane) {
    const uint4* p = (const uint4*)(wfrag + (size_t)fragIdx * 512 + lane * 16);
    b.q[0] = p[0];
    b.q[1] = p[1];
}

// D fragment (f32 16x16) store to row-major [16][128] f16 staging.
// VGPR r: lanes 0-15 -> M=r, lanes 16-31 -> M=r+8, N = lane&15 (+ ncol base).
__device__ __forceinline__ void storeD(v8f acc, _Float16* tile_base, int ncol, int lane) {
    int mbase = (lane >> 4) * 8;
#pragma unroll
    for (int r2 = 0; r2 < 8; ++r2)
        tile_base[(mbase + r2) * 128 + ncol] = (_Float16)acc[r2];
}

// ---------------- pre-pass: (plane, feat, r, r) f32 -> (plane, y, x, feat) f16 ----------------
__global__ void triplane_transpose_kernel(const float* __restrict__ in,
                                          _Float16* __restrict__ out, int r, int total) {
    int i = blockIdx.x * blockDim.x + threadIdx.x;
    if (i >= total) return;
    int ft = i & 15;
    int rest = i >> 4;
    int x = rest % r; rest /= r;
    int y = rest % r;
    int p = rest / r;
    out[i] = (_Float16)in[((p * 16 + ft) * r + y) * r + x];
}

// ---------------- pre-pass: pack weights (f32 row-major [nOut][nIn]) into WMMA B fragments ----
// frag order: w1(16) | w2(32) | wc1(16) | ws(4) | wc2(2)  -> 70 frags total
__global__ void pack_weights_kernel(const float* __restrict__ w1, const float* __restrict__ w2,
                                    const float* __restrict__ wc1, const float* __restrict__ wsw,
                                    const float* __restrict__ wc2, _Float16* __restrict__ frags) {
    int gid = blockIdx.x * blockDim.x + threadIdx.x;
    int frag = gid >> 5;
    int lane = gid & 31;
    if (frag >= 70) return;
    const float* W; int nOut, nIn, ntiles, fo;
    if (frag < 16)      { W = w1;  nOut = 128; nIn = 48;  ntiles = 8; fo = 0;  }
    else if (frag < 48) { W = w2;  nOut = 128; nIn = 128; ntiles = 8; fo = 16; }
    else if (frag < 64) { W = wc1; nOut = 64;  nIn = 128; ntiles = 4; fo = 48; }
    else if (frag < 68) { W = wsw; nOut = 1;   nIn = 128; ntiles = 1; fo = 64; }
    else                { W = wc2; nOut = 3;   nIn = 64;  ntiles = 1; fo = 68; }
    int lf = frag - fo;
    int kt = lf / ntiles, nt = lf % ntiles;
    int n = nt * 16 + (lane & 15);
    int khalf = (lane >> 4) * 8;
    _Float16 vals[16];
#pragma unroll
    for (int e = 0; e < 16; ++e) {
        int v = e >> 1;
        int K = kt * 32 + (v < 4 ? 0 : 16) + khalf + 2 * (v & 3) + (e & 1);
        float f = (n < nOut && K < nIn) ? W[n * nIn + K] : 0.0f;
        vals[e] = (_Float16)f;
    }
    _Float16* dst = frags + (size_t)frag * 512 + lane * 16;
#pragma unroll
    for (int e = 0; e < 16; ++e) dst[e] = vals[e];
}

// ---------------- main kernel: sample triplanes + dual-tile WMMA MLP ----------------
// 128 threads = 4 waves; each wave owns TWO 16-point tiles (32 pts) -> B frags reused 2x.
__launch_bounds__(128)
__global__ void nerf_triplane_main_kernel(const float* __restrict__ coords,
                                          const _Float16* __restrict__ pt0,
                                          const _Float16* __restrict__ pt1,
                                          const _Float16* __restrict__ pt2,
                                          const _Float16* __restrict__ wfrag,
                                          const float* __restrict__ b1,
                                          const float* __restrict__ b2,
                                          const float* __restrict__ bs,
                                          const float* __restrict__ bc1,
                                          const float* __restrict__ bc2,
                                          float* __restrict__ out, int npts) {
    __shared__ _Float16 stage[4 * 2 * 16 * 128]; // 32 KB: [wave][tile][16][128] f16
    const int t = threadIdx.x;
    const int wave = t >> 5;
    const int lane = t & 31;

    // ---------------- triplane sampling: 1 thread per point, packed f16 bilinear --------
    {
        int pl = t; // local point 0..127
        int P = blockIdx.x * 128 + pl;
        if (P >= npts) P = npts - 1;
        float cx = tanhf(coords[P * 3 + 0]);
        float cy = tanhf(coords[P * 3 + 1]);
        float cz = tanhf(coords[P * 3 + 2]);
        float gx[3] = {cx, cy, cx};
        float gy[3] = {cy, cz, cz};
        const _Float16* planes[3] = {pt0, pt1, pt2};
        const int rs[3] = {128, 256, 512};
        _Float16* srow = stage + (pl >> 5) * 4096 + ((pl >> 4) & 1) * 2048 + (pl & 15) * 128;
#pragma unroll
        for (int i = 0; i < 3; ++i) {
            int r = rs[i];
            const _Float16* pt = planes[i];
            v8h accA = splath((_Float16)0.0f);
            v8h accB = splath((_Float16)0.0f);
#pragma unroll
            for (int j = 0; j < 3; ++j) {
                float x = (gx[j] + 1.0f) * 0.5f * (float)(r - 1);
                float y = (gy[j] + 1.0f) * 0.5f * (float)(r - 1);
                float x0f = floorf(x), y0f = floorf(y);
                float wx = x - x0f, wy = y - y0f;
                int x0 = iclamp((int)x0f, 0, r - 1);
                int x1 = iclamp(x0 + 1, 0, r - 1);
                int y0 = iclamp((int)y0f, 0, r - 1);
                int y1 = iclamp(y0 + 1, 0, r - 1);
                const _Float16* base = pt + (size_t)j * r * r * 16;
                const uint4* c00p = (const uint4*)(base + (y0 * r + x0) * 16);
                const uint4* c01p = (const uint4*)(base + (y0 * r + x1) * 16);
                const uint4* c10p = (const uint4*)(base + (y1 * r + x0) * 16);
                const uint4* c11p = (const uint4*)(base + (y1 * r + x1) * 16);
                H8 a00, a01, a10, a11, b00, b01, b10, b11;
                a00.q = c00p[0]; b00.q = c00p[1];
                a01.q = c01p[0]; b01.q = c01p[1];
                a10.q = c10p[0]; b10.q = c10p[1];
                a11.q = c11p[0]; b11.q = c11p[1];
                v8h w00 = splath((_Float16)((1.0f - wx) * (1.0f - wy)));
                v8h w01 = splath((_Float16)(wx * (1.0f - wy)));
                v8h w10 = splath((_Float16)((1.0f - wx) * wy));
                v8h w11 = splath((_Float16)(wx * wy));
                accA = accA + a00.v * w00 + a01.v * w01 + a10.v * w10 + a11.v * w11;
                accB = accB + b00.v * w00 + b01.v * w01 + b10.v * w10 + b11.v * w11;
            }
            H8 oa, ob;
            oa.v = accA;
            ob.v = accB;
            *(uint4*)(srow + i * 16)     = oa.q;
            *(uint4*)(srow + i * 16 + 8) = ob.q;
        }
        uint4 z; z.x = 0; z.y = 0; z.z = 0; z.w = 0;
        *(uint4*)(srow + 48) = z; // zero-pad K = 48..63
        *(uint4*)(srow + 56) = z;
    }
    __syncthreads();

    _Float16* wb0 = stage + wave * 4096;        // tile 0 staging [16][128]
    _Float16* wb1 = wb0 + 2048;                 // tile 1 staging
    const int n0 = lane & 15;

    // ---------------- layer 1: feats[16x48(pad64)] @ w1^T -> h[16x128], lrelu ----------
    {
        FragU a00, a01, a10, a11;
        loadA(a00, wb0, 0, lane);
        loadA(a01, wb0, 1, lane);
        loadA(a10, wb1, 0, lane);
        loadA(a11, wb1, 1, lane);
#pragma unroll
        for (int nt = 0; nt < 8; ++nt) {
            float bv = b1[nt * 16 + n0];
            v8f p = splat8(bv), q = splat8(bv);
            FragU b;
            loadB(b, wfrag, 0 + nt, lane);
            p = wmma_f16(a00.h, b.h, p);
            q = wmma_f16(a10.h, b.h, q);
            loadB(b, wfrag, 8 + nt, lane);
            p = wmma_f16(a01.h, b.h, p);
            q = wmma_f16(a11.h, b.h, q);
            storeD(lrelu8(p), wb0, nt * 16 + n0, lane);
            storeD(lrelu8(q), wb1, nt * 16 + n0, lane);
        }
    }
    __syncthreads();

    // ---------------- layer 2: h @ w2^T -> h2[16x128], lrelu ----------------------------
    {
        FragU h0[4], h1[4];
#pragma unroll
        for (int kt = 0; kt < 4; ++kt) {
            loadA(h0[kt], wb0, kt, lane);
            loadA(h1[kt], wb1, kt, lane);
        }
#pragma unroll
        for (int nt = 0; nt < 8; ++nt) {
            float bv = b2[nt * 16 + n0];
            v8f p = splat8(bv), q = splat8(bv);
#pragma unroll
            for (int kt = 0; kt < 4; ++kt) {
                FragU b;
                loadB(b, wfrag, 16 + kt * 8 + nt, lane);
                p = wmma_f16(h0[kt].h, b.h, p);
                q = wmma_f16(h1[kt].h, b.h, q);
            }
            storeD(lrelu8(p), wb0, nt * 16 + n0, lane);
            storeD(lrelu8(q), wb1, nt * 16 + n0, lane);
        }
    }
    __syncthreads();

    // ---------------- heads: sigma = h2 @ ws^T ; ch = lrelu(h2 @ wc1^T) -----------------
    v8f sacc0, sacc1;
    {
        FragU g0[4], g1[4];
#pragma unroll
        for (int kt = 0; kt < 4; ++kt) {
            loadA(g0[kt], wb0, kt, lane);
            loadA(g1[kt], wb1, kt, lane);
        }
        // sigma (N padded to 16; only column 0 meaningful)
        float sb = (n0 == 0) ? bs[0] : 0.0f;
        sacc0 = splat8(sb);
        sacc1 = splat8(sb);
#pragma unroll
        for (int kt = 0; kt < 4; ++kt) {
            FragU b;
            loadB(b, wfrag, 64 + kt, lane);
            sacc0 = wmma_f16(g0[kt].h, b.h, sacc0);
            sacc1 = wmma_f16(g1[kt].h, b.h, sacc1);
        }
        // wc1 -> ch[16x64], written to staging cols 0..63
#pragma unroll
        for (int nt = 0; nt < 4; ++nt) {
            float bv = bc1[nt * 16 + n0];
            v8f p = splat8(bv), q = splat8(bv);
#pragma unroll
            for (int kt = 0; kt < 4; ++kt) {
                FragU b;
                loadB(b, wfrag, 48 + kt * 4 + nt, lane);
                p = wmma_f16(g0[kt].h, b.h, p);
                q = wmma_f16(g1[kt].h, b.h, q);
            }
            storeD(lrelu8(p), wb0, nt * 16 + n0, lane);
            storeD(lrelu8(q), wb1, nt * 16 + n0, lane);
        }
    }
    __syncthreads();

    // ---------------- color = ch @ wc2^T (N padded to 16, cols 0..2 valid) --------------
    v8f cacc0, cacc1;
    {
        FragU c00, c01, c10, c11;
        loadA(c00, wb0, 0, lane);
        loadA(c01, wb0, 1, lane);
        loadA(c10, wb1, 0, lane);
        loadA(c11, wb1, 1, lane);
        float cb = (n0 < 3) ? bc2[n0] : 0.0f;
        cacc0 = splat8(cb);
        cacc1 = splat8(cb);
        FragU b;
        loadB(b, wfrag, 68, lane);
        cacc0 = wmma_f16(c00.h, b.h, cacc0);
        cacc1 = wmma_f16(c10.h, b.h, cacc1);
        loadB(b, wfrag, 69, lane);
        cacc0 = wmma_f16(c01.h, b.h, cacc0);
        cacc1 = wmma_f16(c11.h, b.h, cacc1);
    }

    // ---------------- output: [color0..2, sigma] per point ------------------------------
    {
        int Pbase = (blockIdx.x * 4 + wave) * 32;
        int mbase = (lane >> 4) * 8;
#pragma unroll
        for (int r2 = 0; r2 < 8; ++r2) {
            int P0 = Pbase + mbase + r2;       // tile 0
            int P1 = P0 + 16;                  // tile 1
            if (P0 < npts) {
                if (n0 < 3)  out[P0 * 4 + n0] = cacc0[r2];
                if (n0 == 0) out[P0 * 4 + 3] = sacc0[r2];
            }
            if (P1 < npts) {
                if (n0 < 3)  out[P1 * 4 + n0] = cacc1[r2];
                if (n0 == 0) out[P1 * 4 + 3] = sacc1[r2];
            }
        }
    }
}

extern "C" void kernel_launch(void* const* d_in, const int* in_sizes, int n_in,
                              void* d_out, int out_size, void* d_ws, size_t ws_size,
                              hipStream_t stream) {
    const float* coords = (const float*)d_in[0];
    const float* p0  = (const float*)d_in[1];
    const float* p1  = (const float*)d_in[2];
    const float* p2  = (const float*)d_in[3];
    const float* w1  = (const float*)d_in[4];
    const float* b1  = (const float*)d_in[5];
    const float* w2  = (const float*)d_in[6];
    const float* b2  = (const float*)d_in[7];
    const float* wsw = (const float*)d_in[8];
    const float* bs  = (const float*)d_in[9];
    const float* wc1 = (const float*)d_in[10];
    const float* bc1 = (const float*)d_in[11];
    const float* wc2 = (const float*)d_in[12];
    const float* bc2 = (const float*)d_in[13];

    const int npts = in_sizes[0] / 3;

    // workspace layout (f16): transposed planes + packed weight fragments
    char* ws = (char*)d_ws;
    size_t o = 0;
    const size_t sz0 = (size_t)3 * 128 * 128 * 16;
    const size_t sz1 = (size_t)3 * 256 * 256 * 16;
    const size_t sz2 = (size_t)3 * 512 * 512 * 16;
    _Float16* pt0 = (_Float16*)(ws + o); o += sz0 * 2;
    _Float16* pt1 = (_Float16*)(ws + o); o += sz1 * 2;
    _Float16* pt2 = (_Float16*)(ws + o); o += sz2 * 2;
    o = (o + 255) & ~(size_t)255;
    _Float16* wfrag = (_Float16*)(ws + o);

    triplane_transpose_kernel<<<(int)((sz0 + 255) / 256), 256, 0, stream>>>(p0, pt0, 128, (int)sz0);
    triplane_transpose_kernel<<<(int)((sz1 + 255) / 256), 256, 0, stream>>>(p1, pt1, 256, (int)sz1);
    triplane_transpose_kernel<<<(int)((sz2 + 255) / 256), 256, 0, stream>>>(p2, pt2, 512, (int)sz2);
    pack_weights_kernel<<<(70 * 32 + 255) / 256, 256, 0, stream>>>(w1, w2, wc1, wsw, wc2, wfrag);

    int nblocks = (npts + 127) / 128;
    nerf_triplane_main_kernel<<<nblocks, 128, 0, stream>>>(
        coords, pt0, pt1, pt2, wfrag, b1, b2, bs, bc1, bc2, (float*)d_out, npts);
}
